// WindowedAttn_34351148433901
// MI455X (gfx1250) — compile-verified, matrix operand order checked by reading, not run on hardware
//
#include <hip/hip_runtime.h>
#include <hip/hip_bf16.h>
#include <cstdint>
#include <cstddef>

// ---------------------------------------------------------------------------
// Windowed causal attention block for MI455X (gfx1250, wave32, WMMA).
//   qkv = x @ Wqkv + bqkv          (bf16 WMMA, f32 accum) -> Q/K/V bf16
//   attn: flash-style, window=256  (bf16 WMMA, f32 softmax/accum,
//                                   async global->LDS K/V staging)
//   out  = attn @ Wout + bout      (bf16 WMMA, f32 accum) -> f32
// ---------------------------------------------------------------------------

typedef __attribute__((ext_vector_type(16))) __bf16        v16bf;
typedef __attribute__((ext_vector_type(8)))  float         v8f;
typedef __attribute__((ext_vector_type(8)))  unsigned int  v8u;

#define D_MODEL 1024
#define N_HEADS 16
#define D_HEAD  64
#define WINDOW  256
#define BATCH   2
#define SEQ     2048
#define TOK     (BATCH * SEQ)        // 4096 tokens
#define QKV_N   (3 * D_MODEL)        // 3072
#define ATTN_SCALE 0.125f            // 64^-0.5

__device__ __forceinline__ unsigned short f2bf(float f) {
  union { float f; unsigned u; } c; c.f = f;
  unsigned u = c.u;
  u += 0x7fffu + ((u >> 16) & 1u);   // round-to-nearest-even
  return (unsigned short)(u >> 16);
}

#if __has_builtin(__builtin_amdgcn_cvt_pk_bf16_f32)
typedef __attribute__((ext_vector_type(2))) __bf16 v2bf;
__device__ __forceinline__ unsigned pack_bf16(float a, float b) {
  v2bf v = __builtin_amdgcn_cvt_pk_bf16_f32(a, b);
  return __builtin_bit_cast(unsigned, v);
}
#else
__device__ __forceinline__ unsigned pack_bf16(float a, float b) {
  return (unsigned)f2bf(a) | ((unsigned)f2bf(b) << 16);
}
#endif

// K-pair base for 16-bit A/B fragments (wave32):
// VGPR r holds K = {kp, kp+1}; lanes 16-31 are offset by 8 in K.
__device__ __forceinline__ int kpair(int r, int half) {
  return 2 * r + ((r >= 4) ? 8 : 0) + 8 * half;
}

__device__ __forceinline__ v8f vzero8() {
  v8f v;
#pragma unroll
  for (int i = 0; i < 8; i++) v[i] = 0.0f;
  return v;
}

__device__ __forceinline__ v8f wmma_bf16(v16bf a, v16bf b, v8f c) {
  // (neg_a, A, neg_b, B, c_mod, C, reuse_a, reuse_b)
  return __builtin_amdgcn_wmma_f32_16x16x32_bf16(false, a, false, b,
                                                 (short)0, c, false, false);
}

// Async global -> LDS copy of 16 bytes (CDNA5 GLOBAL_LOAD_ASYNC_TO_LDS_B128).
// lds_off = byte offset (low 32 bits of generic shared pointer = LDS address).
__device__ __forceinline__ void async_copy16(unsigned lds_off, const void* gptr) {
  unsigned long long ga = (unsigned long long)(uintptr_t)gptr;
  asm volatile("global_load_async_to_lds_b128 %0, %1, off"
               :: "v"(lds_off), "v"(ga) : "memory");
}
__device__ __forceinline__ void async_wait0() {
  asm volatile("s_wait_asynccnt 0x0" ::: "memory");
}

// ---------------------------------------------------------------------------
// GEMM: C[M,N] = A[M,K] (f32) * B[K,N] (f32) + bias[N]
//   mode 0: store f32 row-major into Cf
//   mode 1: store bf16 split into Q/K/V (B,H,S,D_HEAD) at qkvOut
// Block: 256 threads (8 waves, 4x2), tile 128x64, K-step 32.
// B is staged transposed in LDS so every fragment pair is a b32 load.
// ---------------------------------------------------------------------------
__global__ void __launch_bounds__(256)
gemm_bias_kernel(const float* __restrict__ A, const float* __restrict__ B,
                 const float* __restrict__ bias, float* __restrict__ Cf,
                 unsigned short* __restrict__ qkvOut,
                 int M, int N, int K, int mode)
{
  __shared__ __align__(16) unsigned short As[128][34];   // [m][k] bf16, padded
  __shared__ __align__(16) unsigned short Bst[64][34];   // [n][k] bf16 transposed

  const int tid  = threadIdx.x;
  const int lane = tid & 31;
  const int wave = tid >> 5;
  const int half = lane >> 4;
  const int l16  = lane & 15;
  const int wm   = wave >> 1;              // 0..3  (M direction, 32 rows each)
  const int wn   = wave & 1;               // 0..1  (N direction, 32 cols each)
  const int m0   = blockIdx.y * 128;
  const int n0   = blockIdx.x * 64;

  v8f acc[2][2];
#pragma unroll
  for (int i = 0; i < 2; i++)
#pragma unroll
    for (int j = 0; j < 2; j++) acc[i][j] = vzero8();

  const int arow = tid >> 1, acb = (tid & 1) * 16;  // A: 2 thr/row, 16 f32 each
  const int brow = tid >> 3, bcb = (tid & 7) * 8;   // B: 8 thr/row, 8 f32 each

  for (int k0 = 0; k0 < K; k0 += 32) {
    // ---- stage tiles into LDS with f32 -> bf16 conversion ----
    {
      const float* ap = A + (size_t)(m0 + arow) * K + k0 + acb;
#pragma unroll
      for (int i = 0; i < 8; i++)
        *(unsigned*)&As[arow][acb + 2 * i] = pack_bf16(ap[2 * i], ap[2 * i + 1]);
      const float* bp = B + (size_t)(k0 + brow) * N + n0 + bcb;
#pragma unroll
      for (int i = 0; i < 8; i++) Bst[bcb + i][brow] = f2bf(bp[i]);
    }
    __syncthreads();

    if (k0 + 32 < K) {  // hint next tiles into cache (global_prefetch_b8)
      __builtin_prefetch(A + (size_t)(m0 + arow) * K + k0 + 32 + acb, 0, 0);
      __builtin_prefetch(B + (size_t)(k0 + 32 + brow) * N + n0 + bcb, 0, 0);
    }

    // ---- load fragments per documented wave32 layouts (all b32 pairs) ----
    v16bf af[2], bf[2];
#pragma unroll
    for (int i = 0; i < 2; i++) {
      v8u u;
      const int mr = wm * 32 + i * 16 + l16;
#pragma unroll
      for (int r = 0; r < 8; r++)
        u[r] = *(const unsigned*)&As[mr][kpair(r, half)];
      af[i] = __builtin_bit_cast(v16bf, u);
    }
#pragma unroll
    for (int j = 0; j < 2; j++) {
      v8u u;
      const int nc = wn * 32 + j * 16 + l16;
#pragma unroll
      for (int r = 0; r < 8; r++)
        u[r] = *(const unsigned*)&Bst[nc][kpair(r, half)];
      bf[j] = __builtin_bit_cast(v16bf, u);
    }

#pragma unroll
    for (int i = 0; i < 2; i++)
#pragma unroll
      for (int j = 0; j < 2; j++)
        acc[i][j] = wmma_bf16(af[i], bf[j], acc[i][j]);
    __syncthreads();
  }

  // ---- epilogue: bias + store (C/D layout: lane=N, VGPR r -> M = r+8*half) ----
#pragma unroll
  for (int i = 0; i < 2; i++) {
#pragma unroll
    for (int j = 0; j < 2; j++) {
      const int n  = n0 + wn * 32 + j * 16 + l16;
      const float bv = bias[n];
#pragma unroll
      for (int r = 0; r < 8; r++) {
        const int m = m0 + wm * 32 + i * 16 + r + 8 * half;
        const float val = acc[i][j][r] + bv;
        if (mode == 0) {
          Cf[(size_t)m * N + n] = val;
        } else {
          // split (token m, col n) -> which(q/k/v), head, d; (B,H,S,D) bf16
          const int bb = m >> 11, s = m & 2047;
          const int which = n >> 10;
          const int h = (n & 1023) >> 6;
          const int d = n & 63;
          const size_t off =
              (((size_t)which * BATCH * N_HEADS + (size_t)bb * N_HEADS + h) * SEQ
               + (size_t)s) * D_HEAD + d;
          qkvOut[off] = f2bf(val);
        }
      }
    }
  }
}

// ---------------------------------------------------------------------------
// Windowed flash attention.
// Grid: (SEQ/64, N_HEADS, BATCH); block = 128 threads = 4 waves.
// Each wave owns 16 query rows; keys swept in block-uniform 32-wide chunks
// over [max(0,q_blk-256)&~31 , q_blk+63]. K/V chunks (4KB each, contiguous
// in global since layout is (B,H,S,64) bf16) are staged into LDS once per
// block via async global->LDS b128 copies and shared by all 4 waves.
// ---------------------------------------------------------------------------
__global__ void __launch_bounds__(128)
attn_kernel(const unsigned short* __restrict__ Qb,
            const unsigned short* __restrict__ Kb,
            const unsigned short* __restrict__ Vb,
            float* __restrict__ Out)
{
  __shared__ __align__(16) unsigned short Kl[32][64];   // key chunk, [j][d]
  __shared__ __align__(16) unsigned short Vl[32][64];   // val chunk, [j][d]
  __shared__ unsigned short Pl[4][16][34];              // per-wave P (16x32 bf16)

  const int tid  = threadIdx.x;
  const int lane = tid & 31;
  const int w    = tid >> 5;
  const int half = lane >> 4;
  const int l16  = lane & 15;
  const int h    = blockIdx.y;
  const int b    = blockIdx.z;
  const int qblk = blockIdx.x * 64;
  const int q0   = qblk + w * 16;
  const size_t headBase = ((size_t)b * N_HEADS + h) * SEQ * D_HEAD;

  const unsigned kl0 = (unsigned)(uintptr_t)&Kl[0][0];
  const unsigned vl0 = (unsigned)(uintptr_t)&Vl[0][0];

  // Q fragments (A-layout), one per 32-wide d-chunk; contiguous K-pairs.
  v16bf qf[2];
#pragma unroll
  for (int dc = 0; dc < 2; dc++) {
    v8u u;
    const unsigned short* qp = Qb + headBase + (size_t)(q0 + l16) * D_HEAD + dc * 32;
#pragma unroll
    for (int r = 0; r < 8; r++)
      u[r] = *(const unsigned int*)(qp + kpair(r, half));
    qf[dc] = __builtin_bit_cast(v16bf, u);
  }

  v8f o[4];
#pragma unroll
  for (int nb = 0; nb < 4; nb++) o[nb] = vzero8();
  float mrow[8], lrow[8];
#pragma unroll
  for (int r = 0; r < 8; r++) { mrow[r] = -__builtin_inff(); lrow[r] = 0.0f; }

  // Block-uniform chunk range so __syncthreads() is safe for all 4 waves.
  int js = qblk - WINDOW; if (js < 0) js = 0; js &= ~31;
  const int jend = qblk + 63;

  for (int jb = js; jb <= jend; jb += 32) {
    // ---- async stage K/V chunk (2 x 4KB, linear) into LDS ----
    {
      const unsigned short* kg = Kb + headBase + (size_t)jb * D_HEAD;
      const unsigned short* vg = Vb + headBase + (size_t)jb * D_HEAD;
#pragma unroll
      for (int c = 0; c < 2; c++) {
        const int eo = (tid + c * 128) * 8;       // ushort offset, 16B per xfer
        async_copy16(kl0 + eo * 2, kg + eo);
        async_copy16(vl0 + eo * 2, vg + eo);
      }
      async_wait0();
      __syncthreads();
    }

    // ---- scores: S = Q * K^T over two 16-key tiles, contraction over d ----
    v8f sc[2];
#pragma unroll
    for (int t = 0; t < 2; t++) {
      sc[t] = vzero8();
      const int jr = t * 16 + l16;                // row within chunk
#pragma unroll
      for (int dc = 0; dc < 2; dc++) {
        v8u u;
#pragma unroll
        for (int r = 0; r < 8; r++)
          u[r] = *(const unsigned int*)&Kl[jr][dc * 32 + kpair(r, half)];
        sc[t] = wmma_bf16(qf[dc], __builtin_bit_cast(v16bf, u), sc[t]);
      }
    }

    // ---- mask + online softmax (row stats via width-16 shuffles) ----
    float p0v[8], p1v[8];
#pragma unroll
    for (int r = 0; r < 8; r++) {
      const int i = q0 + r + 8 * half;
      float sv[2];
#pragma unroll
      for (int t = 0; t < 2; t++) {
        const int j = jb + t * 16 + l16;
        const bool ok = (j <= i) && (i - j <= WINDOW);
        sv[t] = ok ? sc[t][r] * ATTN_SCALE : -__builtin_inff();
      }
      float cm = fmaxf(sv[0], sv[1]);
#pragma unroll
      for (int off = 1; off < 16; off <<= 1)
        cm = fmaxf(cm, __shfl_xor(cm, off, 16));
      const float mn = fmaxf(mrow[r], cm);
      const float alpha =
          (mn == -__builtin_inff()) ? 1.0f : __expf(mrow[r] - mn);
      const float p0 = (sv[0] == -__builtin_inff()) ? 0.0f : __expf(sv[0] - mn);
      const float p1 = (sv[1] == -__builtin_inff()) ? 0.0f : __expf(sv[1] - mn);
      float rs = p0 + p1;
#pragma unroll
      for (int off = 1; off < 16; off <<= 1)
        rs += __shfl_xor(rs, off, 16);
      lrow[r] = lrow[r] * alpha + rs;
      mrow[r] = mn;
#pragma unroll
      for (int nb = 0; nb < 4; nb++) o[nb][r] *= alpha;
      p0v[r] = p0; p1v[r] = p1;
    }

    // ---- C-layout P -> LDS (row-major 16x32 bf16) ----
#pragma unroll
    for (int r = 0; r < 8; r++) {
      const int mr = r + 8 * half;
      Pl[w][mr][l16]      = f2bf(p0v[r]);
      Pl[w][mr][16 + l16] = f2bf(p1v[r]);
    }
    __syncthreads();

    // ---- reload P as A-fragment; V as B-fragments; accumulate O ----
    v8u pu;
#pragma unroll
    for (int r = 0; r < 8; r++) {
      const int kk = kpair(r, half);
      unsigned lo = Pl[w][l16][kk], hi = Pl[w][l16][kk + 1];
      pu[r] = lo | (hi << 16);
    }
    const v16bf pf = __builtin_bit_cast(v16bf, pu);
#pragma unroll
    for (int nb = 0; nb < 4; nb++) {
      v8u u;
      const int n = nb * 16 + l16;
#pragma unroll
      for (int r = 0; r < 8; r++) {
        const int kk = kpair(r, half);
        unsigned lo = Vl[kk][n], hi = Vl[kk + 1][n];
        u[r] = lo | (hi << 16);
      }
      o[nb] = wmma_bf16(pf, __builtin_bit_cast(v16bf, u), o[nb]);
    }
    __syncthreads();  // protect Kl/Vl/Pl before next chunk's staging
  }

  // ---- finalize: divide by row sums, write f32 (B,S,H*64+d) ----
#pragma unroll
  for (int nb = 0; nb < 4; nb++) {
#pragma unroll
    for (int r = 0; r < 8; r++) {
      const int i = q0 + r + 8 * half;
      const float inv = 1.0f / lrow[r];
      Out[((size_t)b * SEQ + i) * D_MODEL + h * D_HEAD + nb * 16 + l16] =
          o[nb][r] * inv;
    }
  }
}

// ---------------------------------------------------------------------------
extern "C" void kernel_launch(void* const* d_in, const int* in_sizes, int n_in,
                              void* d_out, int out_size, void* d_ws, size_t ws_size,
                              hipStream_t stream) {
  (void)in_sizes; (void)n_in; (void)out_size; (void)ws_size;
  const float* x    = (const float*)d_in[0];
  const float* Wqkv = (const float*)d_in[1];
  const float* bqkv = (const float*)d_in[2];
  const float* Wout = (const float*)d_in[3];
  const float* bout = (const float*)d_in[4];
  float* out = (float*)d_out;

  // Workspace layout: [Q|K|V bf16, 8MB each][attn_out f32, 16MB] = 40MB
  unsigned short* qkv = (unsigned short*)d_ws;
  const size_t QSZ = (size_t)BATCH * N_HEADS * SEQ * D_HEAD;  // elems per tensor
  float* attn_out = (float*)((char*)d_ws + 3 * QSZ * sizeof(unsigned short));

  // 1) QKV projection -> bf16 Q/K/V in (B,H,S,D_HEAD)
  gemm_bias_kernel<<<dim3(QKV_N / 64, TOK / 128), 256, 0, stream>>>(
      x, Wqkv, bqkv, nullptr, qkv, TOK, QKV_N, D_MODEL, 1);

  // 2) windowed flash attention -> f32 (B,S,D_MODEL)
  attn_kernel<<<dim3(SEQ / 64, N_HEADS, BATCH), 128, 0, stream>>>(
      qkv, qkv + QSZ, qkv + 2 * QSZ, attn_out);

  // 3) output projection -> f32 result
  gemm_bias_kernel<<<dim3(D_MODEL / 64, TOK / 128), 256, 0, stream>>>(
      attn_out, Wout, bout, out, nullptr, TOK, D_MODEL, D_MODEL, 0);
}
